// BertNER_57844619543177
// MI455X (gfx1250) — compile-verified, matrix operand order checked by reading, not run on hardware
//
#include <hip/hip_runtime.h>
#include <hip/hip_bf16.h>

// ---- problem dims ----
constexpr int T_LEN = 2048;
constexpr int WD    = 768;    // embedding width
constexpr int HD    = 512;    // lstm hidden
constexpr int G4H   = 2048;   // 4*HD (gates)
constexpr int NCAT  = 5;
constexpr int START_TAG = 3;
constexpr int END_TAG   = 4;
// ---- WMMA tiling ----
constexpr int MT    = T_LEN / 16;   // 128 row tiles of time
constexpr int NT    = G4H / 16;     // 128 col tiles of gates
constexpr int KT_IH = WD / 32;      // 24 K tiles for input proj
constexpr int KT_HH = HD / 32;      // 16 K tiles for recurrence

typedef __attribute__((ext_vector_type(16))) __bf16 v16bf;
typedef __attribute__((ext_vector_type(8)))  float  v8f;
typedef __attribute__((ext_vector_type(4)))  unsigned u32x4;
typedef __attribute__((ext_vector_type(8)))  unsigned u32x8;

// ---- workspace layout (bytes) ----
constexpr size_t OFF_XPACK = 0;                          // 128*24 frags * 1KB = 3 MB
constexpr size_t OFF_WIHP  = OFF_XPACK + 3145728;        // 2*24*128 frags    = 6 MB
constexpr size_t OFF_WHHP  = OFF_WIHP  + 6291456;        // 2*128*16 frags    = 4 MB
constexpr size_t OFF_ZBUF  = OFF_WHHP  + 4194304;        // 2*2048*2048 f32   = 32 MB
constexpr size_t OFF_HOUT  = OFF_ZBUF  + 33554432;       // 2*2048*512 f32    = 8 MB
constexpr size_t OFF_GATES = OFF_HOUT  + 8388608;        // 2 dirs * 2 bufs * 2048 f32
constexpr size_t OFF_BAR   = OFF_GATES + 65536;          // barrier counters
constexpr int    LSTM_SMEM = 262144 + 1024 + 2048;       // Whh slice + h(bf16) + c(f32)

// Position -> K mapping inside a 16-bit WMMA fragment register file
// (ISA 7.12.2: 16-bit A 16x32 layout; B 32x16 is the K-symmetric mirror).
__device__ __forceinline__ int kidx(int j, int hi) {
  const int v = j >> 1, h = j & 1, b = hi ? 8 : 0;
  return (v < 4) ? (v * 2 + h + b) : (16 + (v - 4) * 2 + h + b);
}

__device__ __forceinline__ float sig(float x) { return 1.0f / (1.0f + __expf(-x)); }

// ---------------- init: zero barrier counters (re-run every call) ----------
__global__ void init_kernel(unsigned* bar) {
  if (threadIdx.x < 64) bar[threadIdx.x] = 0u;
}

// ---------------- pack x = emb[sent] into WMMA A fragments (bf16) ----------
__global__ void pack_x_kernel(const int* __restrict__ sent, const float* __restrict__ emb,
                              __bf16* __restrict__ xpack) {
  const int idx = blockIdx.x * 256 + threadIdx.x;          // [mt][kt][lane][j]
  if (idx >= MT * KT_IH * 512) return;
  const int mt = idx / (KT_IH * 512);
  const int r  = idx % (KT_IH * 512);
  const int kt = r / 512;
  const int e  = r % 512;
  const int lane = e >> 4, j = e & 15;
  const int t = mt * 16 + (lane & 15);
  const int k = kt * 32 + kidx(j, lane >> 4);
  xpack[idx] = (__bf16)emb[(size_t)sent[t] * WD + k];
}

// ---------------- pack Wih^T (both dirs) into WMMA B fragments -------------
__global__ void pack_wih_kernel(const float* __restrict__ Wf, const float* __restrict__ Wb,
                                __bf16* __restrict__ wihp) {
  const int idx = blockIdx.x * 256 + threadIdx.x;          // [dir][kt][nt][lane][j]
  if (idx >= 2 * KT_IH * NT * 512) return;
  const int dir = idx / (KT_IH * NT * 512);
  const int r   = idx % (KT_IH * NT * 512);
  const int kt  = r / (NT * 512);
  const int r2  = r % (NT * 512);
  const int nt  = r2 / 512;
  const int e   = r2 % 512;
  const int lane = e >> 4, j = e & 15;
  const int n = nt * 16 + (lane & 15);
  const int k = kt * 32 + kidx(j, lane >> 4);
  const float* W = dir ? Wb : Wf;                          // B[k][n] = W[n][k]
  wihp[idx] = (__bf16)W[(size_t)n * WD + k];
}

// ---------------- pack Whh^T into B fragments, [dir][nt][kt] for LDS slices
__global__ void pack_whh_kernel(const float* __restrict__ Wf, const float* __restrict__ Wb,
                                __bf16* __restrict__ whhp) {
  const int idx = blockIdx.x * 256 + threadIdx.x;          // [dir][nt][kt][lane][j]
  if (idx >= 2 * NT * KT_HH * 512) return;
  const int dir = idx / (NT * KT_HH * 512);
  const int r   = idx % (NT * KT_HH * 512);
  const int nt  = r / (KT_HH * 512);
  const int r2  = r % (KT_HH * 512);
  const int kt  = r2 / 512;
  const int e   = r2 % 512;
  const int lane = e >> 4, j = e & 15;
  const int n = nt * 16 + (lane & 15);
  const int k = kt * 32 + kidx(j, lane >> 4);
  const float* W = dir ? Wb : Wf;
  whhp[idx] = (__bf16)W[(size_t)n * HD + k];
}

// ---------------- z = x @ Wih^T + (bih + bhh), both dirs, WMMA bf16 --------
__global__ void zin_gemm_kernel(const __bf16* __restrict__ xpack, const __bf16* __restrict__ wihp,
                                const float* __restrict__ bih_f, const float* __restrict__ bhh_f,
                                const float* __restrict__ bih_b, const float* __restrict__ bhh_b,
                                float* __restrict__ zbuf) {
  const int w    = blockIdx.x * 8 + (threadIdx.x >> 5);    // one wave = one 16x16 tile
  const int lane = threadIdx.x & 31;
  const int dir  = w >> 14;
  const int mt   = (w >> 7) & 127;
  const int nt   = w & 127;
  const __bf16* ap = xpack + (size_t)mt * KT_IH * 512 + lane * 16;
  const __bf16* bp = wihp + ((size_t)dir * KT_IH * NT + nt) * 512 + lane * 16;
  v8f acc = {};
  for (int kt = 0; kt < KT_IH; ++kt) {
    const v16bf a = *(const v16bf*)(ap + (size_t)kt * 512);
    const v16bf b = *(const v16bf*)(bp + (size_t)kt * NT * 512);
    if (kt + 1 < KT_IH) __builtin_prefetch(bp + (size_t)(kt + 1) * NT * 512, 0, 3);
    acc = __builtin_amdgcn_wmma_f32_16x16x32_bf16(false, a, false, b, (short)0, acc, false, false);
  }
  const int col = nt * 16 + (lane & 15);
  const float bias = (dir ? bih_b[col] + bhh_b[col] : bih_f[col] + bhh_f[col]);
  const int hi = lane >> 4;                                // C/D: vgpr v holds row v + 8*hi
  const size_t base = ((size_t)dir * T_LEN + (size_t)mt * 16 + hi * 8) * G4H + col;
#pragma unroll
  for (int v = 0; v < 8; ++v) zbuf[base + (size_t)v * G4H] = acc[v] + bias;
}

// ---------------- persistent BiLSTM recurrence -----------------------------
// 16 blocks: dir = blk/8, each dir split over 8 WGs; Whh slice LDS-resident,
// staged by the Tensor Data Mover (TDM) in a single descriptor.
__global__ void lstm_kernel(const float* __restrict__ zbuf, const __bf16* __restrict__ whhp,
                            float* __restrict__ hout, float* __restrict__ gates,
                            unsigned* __restrict__ bar) {
  extern __shared__ char smem[];
  __bf16* whh_lds = (__bf16*)smem;                         // 256 KB: 16 nt x 16 kt frags
  __bf16* hbf     = (__bf16*)(smem + 262144);              // h as bf16 (A operand)
  float*  cbuf    = (float*)(smem + 262144 + 1024);        // cell state (replicated/WG)

  const int tid  = threadIdx.x;
  const int dir  = blockIdx.x >> 3;
  const int wg   = blockIdx.x & 7;
  const int wave = tid >> 5;
  const int lane = tid & 31;

  // ---- TDM: DMA this WG's contiguous 256 KB Whh slice into LDS -----------
  if (tid < 32) {  // one wave issues the descriptor; TENSORcnt tracks it
    const unsigned long long ga =
        (unsigned long long)(const void*)(whhp + (size_t)(dir * NT + wg * 16) * KT_HH * 512);
    const unsigned lds_addr = (unsigned)(size_t)whh_lds;   // generic LDS addr[31:0]
    u32x4 g0;
    g0[0] = 1u;                                            // count=1, user D#, no gather
    g0[1] = lds_addr;                                      // D#.lds_addr
    g0[2] = (unsigned)(ga & 0xffffffffu);                  // global_addr[31:0]
    g0[3] = (unsigned)((ga >> 32) & 0x01ffffffu) | (2u << 30);  // addr[56:32] | type=2
    u32x8 g1;
    g1[0] = (3u << 16);                                    // mask=0, data_size=3 (8B)
    g1[1] = 0x8000u << 16;                                 // tensor_dim0 = 32768 (8B units)
    g1[2] = 1u << 16;                                      // tensor_dim1 = 1
    g1[3] = 0x8000u << 16;                                 // tile_dim0 = 32768
    g1[4] = 1u;                                            // tile_dim1 = 1
    g1[5] = 0x8000u;                                       // tensor_dim0_stride = 32768
    g1[6] = 0u;
    g1[7] = 0u;
    asm volatile("tensor_load_to_lds %0, %1" :: "s"(g0), "s"(g1) : "memory");
    __builtin_amdgcn_s_wait_tensorcnt(0);
  }
  for (int i = tid; i < HD; i += 256) { hbf[i] = (__bf16)0.0f; cbuf[i] = 0.0f; }
  __syncthreads();

  unsigned* cnt = bar + dir * 32;                          // per-dir counter, own cacheline

  for (int t = 0; t < T_LEN; ++t) {
    const int tt = dir ? (T_LEN - 1 - t) : t;
    const size_t zrow = ((size_t)dir * T_LEN + (size_t)tt) * G4H;
    float* gptr = gates + (size_t)(((dir << 1) | (t & 1))) * G4H;   // double-buffered

    // produce: gates[n] = z[tt][n] + (h @ Whh^T)[n] via M=1-padded WMMA
#pragma unroll
    for (int q = 0; q < 2; ++q) {
      const int ntl = wave * 2 + q;                        // 0..15 local col tile
      const __bf16* bb = whh_lds + (size_t)ntl * (KT_HH * 512) + lane * 16;
      v8f acc = {};
      for (int kt = 0; kt < KT_HH; ++kt) {
        union { v16bf v; uint4 u[2]; } a;
        a.u[0] = make_uint4(0u, 0u, 0u, 0u);
        a.u[1] = make_uint4(0u, 0u, 0u, 0u);
        if ((lane & 15) == 0) {                            // only row M==0 carries h
          const int kb = kt * 32 + ((lane >> 4) << 3);
          a.u[0] = *(const uint4*)(hbf + kb);              // K = kb..kb+7
          a.u[1] = *(const uint4*)(hbf + kb + 16);         // K = kb+16..kb+23
        }
        const v16bf b = *(const v16bf*)(bb + (size_t)kt * 512);
        acc = __builtin_amdgcn_wmma_f32_16x16x32_bf16(false, a.v, false, b, (short)0, acc,
                                                      false, false);
      }
      if (lane < 16) {                                     // D row 0 lives in acc[0], lanes 0-15
        const int col = (wg * 16 + ntl) * 16 + lane;
        gptr[col] = acc[0] + zbuf[zrow + col];
      }
    }

    // device-scope sync of the 8 WGs of this direction (cluster barrier if
    // cluster-dispatched; atomic arrive/spin otherwise).
    __threadfence();
    __syncthreads();
    if (tid == 0) {
      if (__builtin_amdgcn_cluster_id_x() != 0) {
        __builtin_amdgcn_s_cluster_barrier();
      } else {
        atomicAdd(cnt, 1u);
        const unsigned target = (unsigned)(t + 1) * 8u;
        while (__hip_atomic_load(cnt, __ATOMIC_RELAXED, __HIP_MEMORY_SCOPE_AGENT) < target)
          __builtin_amdgcn_s_sleep(1);
      }
    }
    __syncthreads();
    __threadfence();                                       // acquire other WGs' gate writes

    // consume: every WG computes the full cell update so h stays LDS-local
    for (int cell = tid; cell < HD; cell += 256) {
      const float gi = gptr[cell];
      const float gf = gptr[HD + cell];
      const float gg = gptr[2 * HD + cell];
      const float go = gptr[3 * HD + cell];
      const float c  = sig(gf) * cbuf[cell] + sig(gi) * tanhf(gg);
      const float h  = sig(go) * tanhf(c);
      cbuf[cell] = c;
      hbf[cell]  = (__bf16)h;
      if (wg == 0) hout[((size_t)dir * T_LEN + tt) * HD + cell] = h;
    }
    __syncthreads();
  }
}

// ---------------- FC + Viterbi decode (single block) -----------------------
__global__ void fc_viterbi_kernel(const float* __restrict__ hout, const float* __restrict__ fcw,
                                  const float* __restrict__ fcb, const float* __restrict__ trans,
                                  float* __restrict__ out) {
  __shared__ float sfeat[T_LEN * NCAT];                    // 40 KB
  __shared__ unsigned char sparent[(T_LEN - 1) * NCAT];    // ~10 KB
  const int tid = threadIdx.x;

  for (int t = tid; t < T_LEN; t += 1024) {
    const float* hf = hout + (size_t)t * HD;
    const float* hb = hout + ((size_t)T_LEN + t) * HD;
    float acc[NCAT];
#pragma unroll
    for (int c = 0; c < NCAT; ++c) acc[c] = fcb[c];
    for (int j = 0; j < HD; ++j) {
      const float a = hf[j];
#pragma unroll
      for (int c = 0; c < NCAT; ++c) acc[c] += a * fcw[c * (2 * HD) + j];
    }
    for (int j = 0; j < HD; ++j) {
      const float a = hb[j];
#pragma unroll
      for (int c = 0; c < NCAT; ++c) acc[c] += a * fcw[c * (2 * HD) + HD + j];
    }
#pragma unroll
    for (int c = 0; c < NCAT; ++c) sfeat[t * NCAT + c] = acc[c];
  }
  __syncthreads();

  if (tid < 32) {                                          // wave0: lane i == tag i
    const int lane = tid;
    const int i = (lane < NCAT) ? lane : 0;
    float tri[NCAT];
#pragma unroll
    for (int j = 0; j < NCAT; ++j) tri[j] = trans[i * NCAT + j];
    float layer = sfeat[i] + trans[i * NCAT + START_TAG];
    for (int t = 1; t < T_LEN; ++t) {
      const float ft = sfeat[t * NCAT + i];
      float best = -3.0e38f; int bj = 0;
#pragma unroll
      for (int j = 0; j < NCAT; ++j) {
        const float lj = __shfl(layer, j, 32);
        const float s = ft + tri[j] + lj;
        if (s > best) { best = s; bj = j; }                // strict > == first-max (argmax tie)
      }
      if (lane < NCAT) sparent[(t - 1) * NCAT + lane] = (unsigned char)bj;
      layer = best;
    }
    const float fin = layer + trans[END_TAG * NCAT + i];
    float bestf = -3.0e38f; int bi = 0;
#pragma unroll
    for (int j = 0; j < NCAT; ++j) {
      const float fj = __shfl(fin, j, 32);
      if (fj > bestf) { bestf = fj; bi = j; }
    }
    if (lane == 0) {
      out[T_LEN] = bestf;
      int idx = bi;
      out[T_LEN - 1] = (float)idx;
      for (int k = T_LEN - 2; k >= 0; --k) {
        idx = sparent[k * NCAT + idx];
        out[k] = (float)idx;
      }
    }
  }
}

extern "C" void kernel_launch(void* const* d_in, const int* in_sizes, int n_in,
                              void* d_out, int out_size, void* d_ws, size_t ws_size,
                              hipStream_t stream) {
  (void)in_sizes; (void)n_in; (void)out_size; (void)ws_size;
  const int*   sent  = (const int*)d_in[0];
  const float* emb   = (const float*)d_in[1];
  const float* Wih_f = (const float*)d_in[2];
  const float* Whh_f = (const float*)d_in[3];
  const float* bih_f = (const float*)d_in[4];
  const float* bhh_f = (const float*)d_in[5];
  const float* Wih_b = (const float*)d_in[6];
  const float* Whh_b = (const float*)d_in[7];
  const float* bih_b = (const float*)d_in[8];
  const float* bhh_b = (const float*)d_in[9];
  const float* fc_w  = (const float*)d_in[10];
  const float* fc_b  = (const float*)d_in[11];
  const float* trans = (const float*)d_in[12];

  char* ws = (char*)d_ws;
  __bf16*  xpack = (__bf16*)(ws + OFF_XPACK);
  __bf16*  wihp  = (__bf16*)(ws + OFF_WIHP);
  __bf16*  whhp  = (__bf16*)(ws + OFF_WHHP);
  float*   zbuf  = (float*)(ws + OFF_ZBUF);
  float*   hout  = (float*)(ws + OFF_HOUT);
  float*   gates = (float*)(ws + OFF_GATES);
  unsigned* bar  = (unsigned*)(ws + OFF_BAR);

  hipFuncSetAttribute((const void*)lstm_kernel,
                      hipFuncAttributeMaxDynamicSharedMemorySize, LSTM_SMEM);

  init_kernel<<<1, 64, 0, stream>>>(bar);
  pack_x_kernel<<<(MT * KT_IH * 512) / 256, 256, 0, stream>>>(sent, emb, xpack);
  pack_wih_kernel<<<(2 * KT_IH * NT * 512) / 256, 256, 0, stream>>>(Wih_f, Wih_b, wihp);
  pack_whh_kernel<<<(2 * NT * KT_HH * 512) / 256, 256, 0, stream>>>(Whh_f, Whh_b, whhp);
  zin_gemm_kernel<<<(2 * MT * NT) / 8, 256, 0, stream>>>(xpack, wihp, bih_f, bhh_f, bih_b,
                                                         bhh_b, zbuf);
  lstm_kernel<<<16, 256, LSTM_SMEM, stream>>>(zbuf, whhp, hout, gates, bar);
  fc_viterbi_kernel<<<1, 1024, 0, stream>>>(hout, fc_w, fc_b, trans, (float*)d_out);
}
